// TinyMambaClassifier_11613591569109
// MI455X (gfx1250) — compile-verified, matrix operand order checked by reading, not run on hardware
//
#include <hip/hip_runtime.h>
#include <hip/hip_bf16.h>

typedef __attribute__((ext_vector_type(16))) _Float16 v16h;
typedef __attribute__((ext_vector_type(8)))  _Float16 v8h;
typedef __attribute__((ext_vector_type(8)))  float    v8f;
typedef __attribute__((ext_vector_type(4)))  int      v4i;

#define BATCH 4
#define SEQ   513
#define ROWS  (BATCH*SEQ)      // 2052
#define ROWSP 2080             // padded to multiple of 32 (GEMM M-tile)
#define DMODEL 192
#define DINNER 384
#define DSTATE 64
#define DTRANK 12
#define XDBL_N 140
#define XDBL_NP 160            // padded to multiple of 32 (GEMM N-tile)
#define NLAYER 8
#define SPLIT 4                // channel-split of the scan per batch

#if __has_builtin(__builtin_amdgcn_global_load_async_to_lds_b128)
#define ASYNC_LDS 1
#else
#define ASYNC_LDS 0
#endif

#if ASYNC_LDS
typedef __attribute__((address_space(1))) v4i gv4i;   // global int4*
typedef __attribute__((address_space(3))) v4i lv4i;   // LDS int4*
template <int N> __device__ __forceinline__ void wait_async() {
#if __has_builtin(__builtin_amdgcn_s_wait_asynccnt)
    __builtin_amdgcn_s_wait_asynccnt(N);
#else
    asm volatile("s_wait_asynccnt %0" :: "i"(N) : "memory");
#endif
}
#endif

// ---------------------------------------------------------------------------
// Embedding gather + CLS append: x[b,l,:] (f32)
// ---------------------------------------------------------------------------
__global__ void gather_kernel(const int* __restrict__ ids,
                              const float* __restrict__ embed,
                              const float* __restrict__ cls,
                              float* __restrict__ x) {
    int idx = blockIdx.x * blockDim.x + threadIdx.x;
    if (idx >= ROWS * DMODEL) return;
    int rtot = idx / DMODEL, dcol = idx - rtot * DMODEL;
    int b = rtot / SEQ, l = rtot - b * SEQ;
    float v;
    if (l == SEQ - 1) v = cls[dcol];
    else              v = embed[(size_t)ids[b * (SEQ - 1) + l] * DMODEL + dcol];
    x[idx] = v;
}

// ---------------------------------------------------------------------------
// Zero the padded rows (2052..2079) of the f16 activation staging buffers
// ---------------------------------------------------------------------------
__global__ void zero_pads(_Float16* xn_h, _Float16* u_h, _Float16* y_h) {
    int idx = blockIdx.x * blockDim.x + threadIdx.x;
    const int PAD = ROWSP - ROWS; // 28
    if (idx < PAD * DMODEL) { xn_h[(ROWS + idx / DMODEL) * DMODEL + idx % DMODEL] = (_Float16)0.f; return; }
    idx -= PAD * DMODEL;
    if (idx < PAD * DINNER) { u_h[(ROWS + idx / DINNER) * DINNER + idx % DINNER] = (_Float16)0.f; return; }
    idx -= PAD * DINNER;
    if (idx < PAD * DINNER) { y_h[(ROWS + idx / DINNER) * DINNER + idx % DINNER] = (_Float16)0.f; }
}

// ---------------------------------------------------------------------------
// f32 -> f16 weight conversion with optional row padding (zero-filled)
// ---------------------------------------------------------------------------
__global__ void cvt_pad(const float* __restrict__ src, _Float16* __restrict__ dst,
                        int nmat, int rows, int padRows, int cols) {
    int total = nmat * padRows * cols;
    int idx = blockIdx.x * blockDim.x + threadIdx.x;
    if (idx >= total) return;
    int m = idx / (padRows * cols);
    int rem = idx - m * padRows * cols;
    int r = rem / cols, c = rem - r * cols;
    dst[idx] = (r < rows) ? (_Float16)src[((size_t)m * rows + r) * cols + c] : (_Float16)0.f;
}

// ---------------------------------------------------------------------------
// RMSNorm per row (one wave32 per row), writes f16 for the WMMA GEMM
// ---------------------------------------------------------------------------
__global__ void rmsnorm_rows(const float* __restrict__ x, const float* __restrict__ w,
                             _Float16* __restrict__ out) {
    int row = blockIdx.x;
    int lane = threadIdx.x;
    const float* xr = x + (size_t)row * DMODEL;
    float s = 0.f;
    for (int i = lane; i < DMODEL; i += 32) { float v = xr[i]; s += v * v; }
    #pragma unroll
    for (int off = 16; off; off >>= 1) s += __shfl_xor(s, off, 32);
    float inv = rsqrtf(s * (1.0f / DMODEL) + 1e-6f);
    _Float16* o = out + (size_t)row * DMODEL;
    for (int i = lane; i < DMODEL; i += 32) o[i] = (_Float16)(w[i] * xr[i] * inv);
}

// ---------------------------------------------------------------------------
// WMMA GEMM (NT): C[M,N] = A[Mp,K] * B[Np,K]^T, f16 in, f32 accumulate.
// One wave computes a 32x32 tile as 2x2 WMMA tiles: 4 fragment loads feed
// 4 v_wmma_f32_16x16x32_f16 (2x reuse of each fragment -> 16 FLOP/byte).
// Fragment layout per ISA 7.12.2; padded buffers keep EXEC all-1s.
// ---------------------------------------------------------------------------
__device__ __forceinline__ v16h load_frag(const _Float16* p) {
    v8h lo = *(const v8h*)p;
    v8h hi = *(const v8h*)(p + 16);
    v16h f;
    #pragma unroll
    for (int i = 0; i < 8; ++i) { f[i] = lo[i]; f[8 + i] = hi[i]; }
    return f;
}

__device__ __forceinline__ void store_tile(float* __restrict__ C, const v8f& c,
                                           int m0, int n0, int M, int N, int ldc,
                                           int lane, int accumulate) {
    int col = n0 + (lane & 15);
    int rowbase = m0 + (lane >> 4) * 8;
    if (col < N) {
        #pragma unroll
        for (int i = 0; i < 8; ++i) {
            int row = rowbase + i;
            if (row < M) {
                float* cp = C + (size_t)row * ldc + col;
                *cp = accumulate ? (*cp + c[i]) : c[i];
            }
        }
    }
}

__global__ void wmma_gemm_nt(const _Float16* __restrict__ A,
                             const _Float16* __restrict__ B,
                             float* __restrict__ C,
                             int M, int N, int K, int ldc, int accumulate) {
    int m0 = blockIdx.x * 32;
    int n0 = blockIdx.y * 32;
    int lane = threadIdx.x;
    int r = lane & 15;
    int half = lane >> 4;
    const _Float16* a0 = A + (size_t)(m0 + r) * K + half * 8;
    const _Float16* a1 = a0 + (size_t)16 * K;
    const _Float16* b0 = B + (size_t)(n0 + r) * K + half * 8;
    const _Float16* b1 = b0 + (size_t)16 * K;
    v8f c00 = {}, c01 = {}, c10 = {}, c11 = {};
    for (int k0 = 0; k0 < K; k0 += 32) {
        v16h af0 = load_frag(a0 + k0);
        v16h af1 = load_frag(a1 + k0);
        v16h bf0 = load_frag(b0 + k0);
        v16h bf1 = load_frag(b1 + k0);
        c00 = __builtin_amdgcn_wmma_f32_16x16x32_f16(false, af0, false, bf0, (short)0, c00, false, false);
        c01 = __builtin_amdgcn_wmma_f32_16x16x32_f16(false, af0, false, bf1, (short)0, c01, false, false);
        c10 = __builtin_amdgcn_wmma_f32_16x16x32_f16(false, af1, false, bf0, (short)0, c10, false, false);
        c11 = __builtin_amdgcn_wmma_f32_16x16x32_f16(false, af1, false, bf1, (short)0, c11, false, false);
    }
    store_tile(C, c00, m0,      n0,      M, N, ldc, lane, accumulate);
    store_tile(C, c01, m0,      n0 + 16, M, N, ldc, lane, accumulate);
    store_tile(C, c10, m0 + 16, n0,      M, N, ldc, lane, accumulate);
    store_tile(C, c11, m0 + 16, n0 + 16, M, N, ldc, lane, accumulate);
}

// ---------------------------------------------------------------------------
// Depthwise causal conv (K=3) + bias + SiLU; writes f32 (scan) and f16 (GEMM)
// ---------------------------------------------------------------------------
__global__ void conv_silu(const float* __restrict__ xz, const float* __restrict__ cw,
                          const float* __restrict__ cb,
                          float* __restrict__ u_f, _Float16* __restrict__ u_h) {
    int idx = blockIdx.x * blockDim.x + threadIdx.x;
    if (idx >= ROWS * DINNER) return;
    int rtot = idx / DINNER, c = idx - rtot * DINNER;
    int b = rtot / SEQ, l = rtot - b * SEQ;
    const float* w = cw + c * 3;
    float acc = cb[c];
    #pragma unroll
    for (int j = 0; j < 3; ++j) {
        int ll = l - 2 + j;
        if (ll >= 0) acc += w[j] * xz[(size_t)(b * SEQ + ll) * (2 * DINNER) + c];
    }
    float s = acc / (1.f + __expf(-acc));
    u_f[idx] = s;
    u_h[idx] = (_Float16)s;
}

// ---------------------------------------------------------------------------
// delta = softplus(dt @ dpw^T + dpb)    (K=12 -> plain VALU dot product)
// ---------------------------------------------------------------------------
__global__ void delta_kernel(const float* __restrict__ xdbl, const float* __restrict__ dpw,
                             const float* __restrict__ dpb, float* __restrict__ delta) {
    int idx = blockIdx.x * blockDim.x + threadIdx.x;
    if (idx >= ROWS * DINNER) return;
    int rtot = idx / DINNER, d = idx - rtot * DINNER;
    const float* xr = xdbl + (size_t)rtot * XDBL_NP;
    const float* wr = dpw + d * DTRANK;
    float acc = dpb[d];
    #pragma unroll
    for (int j = 0; j < DTRANK; ++j) acc += xr[j] * wr[j];
    delta[idx] = (acc > 20.f) ? acc : log1pf(__expf(acc));
}

// ---------------------------------------------------------------------------
// Selective-state scan. Grid = BATCH*SPLIT blocks x 192 threads; each block
// owns 96 channels, thread = (channel, state-half), h[32]+A[32] in registers.
// B/C rows (128 contiguous floats) double-buffered in LDS; on gfx1250 the
// next row is fetched with one async GLOBAL_LOAD_ASYNC_TO_LDS_B128 by wave 0
// (32 lanes x 16B) overlapped with compute, synchronized via s_wait_asynccnt.
// ---------------------------------------------------------------------------
__global__ void __launch_bounds__(2 * DINNER / SPLIT)
scan_kernel(const float* __restrict__ delta, const float* __restrict__ xdbl,
            const float* __restrict__ u_f, const float* __restrict__ Alog,
            float* __restrict__ y_f) {
    const int CH = DINNER / SPLIT;           // 96 channels per block
    int b    = blockIdx.x / SPLIT;
    int part = blockIdx.x - b * SPLIT;
    int tid  = threadIdx.x;                  // 0..191
    int d    = part * CH + (tid >> 1);       // channel
    int p    = tid & 1;                      // state half
    float h[32], A[32];
    const float* Ar = Alog + (size_t)d * DSTATE + p * 32;
    #pragma unroll
    for (int s = 0; s < 32; ++s) { h[s] = 0.f; A[s] = -__expf(Ar[s]); }

    __shared__ __align__(16) float BC[2][2 * DSTATE];   // [buf][B(64) C(64)]
    const float* row0 = xdbl + (size_t)b * SEQ * XDBL_NP + DTRANK;

#if ASYNC_LDS
    if (tid < 32) {
        // prologue: row 0 -> buffer 0 (each lane moves 16 bytes)
        __builtin_amdgcn_global_load_async_to_lds_b128(
            (gv4i*)(row0 + tid * 4), (lv4i*)&BC[0][tid * 4], 0, 0);
    }
#endif
    for (int l = 0; l < SEQ; ++l) {
        size_t r = (size_t)b * SEQ + l;
        __syncthreads();                      // everyone done with buf[(l-1)&1]
#if ASYNC_LDS
        if (tid < 32) {
            if (l + 1 < SEQ) {
                __builtin_amdgcn_global_load_async_to_lds_b128(
                    (gv4i*)(row0 + (size_t)(l + 1) * XDBL_NP + tid * 4),
                    (lv4i*)&BC[(l + 1) & 1][tid * 4], 0, 0);
                wait_async<1>();              // row l complete
            } else {
                wait_async<0>();
            }
        }
#else
        if (tid < 2 * DSTATE) BC[l & 1][tid] = row0[(size_t)l * XDBL_NP + tid];
#endif
        __syncthreads();                      // row l visible to all waves
        float dv = delta[r * DINNER + d];
        float du = dv * u_f[r * DINNER + d];
        const float* Bp = &BC[l & 1][p * 32];
        const float* Cp = &BC[l & 1][DSTATE + p * 32];
        float y = 0.f;
        #pragma unroll
        for (int s = 0; s < 32; ++s) {
            h[s] = __expf(dv * A[s]) * h[s] + du * Bp[s];
            y += h[s] * Cp[s];
        }
        y += __shfl_xor(y, 1, 32);
        if (p == 0) y_f[r * DINNER + d] = y;
    }
}

// ---------------------------------------------------------------------------
// y = (y + u*D) * silu(z), written as f16 for the out_proj WMMA GEMM
// ---------------------------------------------------------------------------
__global__ void post_kernel(const float* __restrict__ y_f, const float* __restrict__ u_f,
                            const float* __restrict__ Dp, const float* __restrict__ xz,
                            _Float16* __restrict__ y_h) {
    int idx = blockIdx.x * blockDim.x + threadIdx.x;
    if (idx >= ROWS * DINNER) return;
    int rtot = idx / DINNER, c = idx - rtot * DINNER;
    float zv = xz[(size_t)rtot * (2 * DINNER) + DINNER + c];
    float sz = zv / (1.f + __expf(-zv));
    y_h[idx] = (_Float16)((y_f[idx] + u_f[idx] * Dp[c]) * sz);
}

// ---------------------------------------------------------------------------
// Final RMSNorm of last token + classifier head (tiny: 8 outputs)
// ---------------------------------------------------------------------------
__global__ void head_kernel(const float* __restrict__ x, const float* __restrict__ fw,
                            const float* __restrict__ hw, const float* __restrict__ hb,
                            float* __restrict__ out) {
    int tid = threadIdx.x;
    if (tid >= BATCH * 2) return;
    int b = tid >> 1, c = tid & 1;
    const float* xr = x + ((size_t)b * SEQ + (SEQ - 1)) * DMODEL;
    float ms = 0.f;
    for (int i = 0; i < DMODEL; ++i) { float v = xr[i]; ms += v * v; }
    float inv = rsqrtf(ms * (1.0f / DMODEL) + 1e-6f);
    const float* w = hw + c * DMODEL;
    float acc = hb[c];
    for (int i = 0; i < DMODEL; ++i) acc += (fw[i] * xr[i] * inv) * w[i];
    out[b * 2 + c] = acc;
}

// ---------------------------------------------------------------------------
// Host orchestration
// ---------------------------------------------------------------------------
extern "C" void kernel_launch(void* const* d_in, const int* in_sizes, int n_in,
                              void* d_out, int out_size, void* d_ws, size_t ws_size,
                              hipStream_t stream) {
    const int*   ids   = (const int*)d_in[0];
    const float* cls   = (const float*)d_in[1];
    const float* embed = (const float*)d_in[2];
    const float* nws   = (const float*)d_in[3];
    const float* ipw   = (const float*)d_in[4];
    const float* cw    = (const float*)d_in[5];
    const float* cb    = (const float*)d_in[6];
    const float* xpw   = (const float*)d_in[7];
    const float* dpw   = (const float*)d_in[8];
    const float* dpb   = (const float*)d_in[9];
    const float* Alog  = (const float*)d_in[10];
    const float* Dsw   = (const float*)d_in[11];
    const float* opw   = (const float*)d_in[12];
    const float* fnw   = (const float*)d_in[13];
    const float* hw    = (const float*)d_in[14];
    const float* hb    = (const float*)d_in[15];
    float* out = (float*)d_out;

    char* ws = (char*)d_ws;
    size_t off = 0;
    auto alloc = [&](size_t bytes) { size_t o = off; off = (off + bytes + 255) & ~(size_t)255; return o; };
    float*    x     = (float*)   (ws + alloc((size_t)ROWS  * DMODEL  * 4));
    _Float16* xn_h  = (_Float16*)(ws + alloc((size_t)ROWSP * DMODEL  * 2));
    float*    xz    = (float*)   (ws + alloc((size_t)ROWS  * 2*DINNER* 4));
    float*    u_f   = (float*)   (ws + alloc((size_t)ROWS  * DINNER  * 4));
    _Float16* u_h   = (_Float16*)(ws + alloc((size_t)ROWSP * DINNER  * 2));
    float*    xdbl  = (float*)   (ws + alloc((size_t)ROWS  * XDBL_NP * 4));
    float*    delta = (float*)   (ws + alloc((size_t)ROWS  * DINNER  * 4));
    float*    y_f   = (float*)   (ws + alloc((size_t)ROWS  * DINNER  * 4));
    _Float16* y_h   = (_Float16*)(ws + alloc((size_t)ROWSP * DINNER  * 2));
    _Float16* ipw_h = (_Float16*)(ws + alloc((size_t)NLAYER * 2*DINNER * DMODEL * 2));
    _Float16* xpw_h = (_Float16*)(ws + alloc((size_t)NLAYER * XDBL_NP * DINNER * 2));
    _Float16* opw_h = (_Float16*)(ws + alloc((size_t)NLAYER * DMODEL  * DINNER * 2));

    // ---- init: gather embeddings, zero activation pads, convert weights ----
    gather_kernel<<<(ROWS * DMODEL + 255) / 256, 256, 0, stream>>>(ids, embed, cls, x);
    {
        int npad = (ROWSP - ROWS) * (DMODEL + 2 * DINNER);
        zero_pads<<<(npad + 255) / 256, 256, 0, stream>>>(xn_h, u_h, y_h);
    }
    {
        int n1 = NLAYER * 2 * DINNER * DMODEL;
        cvt_pad<<<(n1 + 255) / 256, 256, 0, stream>>>(ipw, ipw_h, NLAYER, 2 * DINNER, 2 * DINNER, DMODEL);
        int n2 = NLAYER * XDBL_NP * DINNER;
        cvt_pad<<<(n2 + 255) / 256, 256, 0, stream>>>(xpw, xpw_h, NLAYER, XDBL_N, XDBL_NP, DINNER);
        int n3 = NLAYER * DMODEL * DINNER;
        cvt_pad<<<(n3 + 255) / 256, 256, 0, stream>>>(opw, opw_h, NLAYER, DMODEL, DMODEL, DINNER);
    }

    const int NE = (ROWS * DINNER + 255) / 256;
    for (int l = 0; l < NLAYER; ++l) {
        rmsnorm_rows<<<ROWS, 32, 0, stream>>>(x, nws + l * DMODEL, xn_h);

        dim3 g1(ROWSP / 32, (2 * DINNER) / 32);
        wmma_gemm_nt<<<g1, 32, 0, stream>>>(xn_h, ipw_h + (size_t)l * 2 * DINNER * DMODEL,
                                            xz, ROWS, 2 * DINNER, DMODEL, 2 * DINNER, 0);

        conv_silu<<<NE, 256, 0, stream>>>(xz, cw + l * DINNER * 3, cb + l * DINNER, u_f, u_h);

        dim3 g2(ROWSP / 32, XDBL_NP / 32);
        wmma_gemm_nt<<<g2, 32, 0, stream>>>(u_h, xpw_h + (size_t)l * XDBL_NP * DINNER,
                                            xdbl, ROWS, XDBL_NP, DINNER, XDBL_NP, 0);

        delta_kernel<<<NE, 256, 0, stream>>>(xdbl, dpw + l * DINNER * DTRANK, dpb + l * DINNER, delta);

        scan_kernel<<<BATCH * SPLIT, 2 * DINNER / SPLIT, 0, stream>>>(
            delta, xdbl, u_f, Alog + (size_t)l * DINNER * DSTATE, y_f);

        post_kernel<<<NE, 256, 0, stream>>>(y_f, u_f, Dsw + l * DINNER, xz, y_h);

        dim3 g3(ROWSP / 32, DMODEL / 32);
        wmma_gemm_nt<<<g3, 32, 0, stream>>>(y_h, opw_h + (size_t)l * DMODEL * DINNER,
                                            x, ROWS, DMODEL, DINNER, DMODEL, 1);
    }

    head_kernel<<<1, 32, 0, stream>>>(x, fnw, hw, hb, out);
}
